// Hyperbolic_Lorentz_Prototype_Distance_Layer_67946382623309
// MI455X (gfx1250) — matrix operand order repaired and checked by reading, not compiled, not run on hardware
//
#include <hip/hip_runtime.h>
#include <hip/hip_bf16.h>
#include <math.h>

// ---------------------------------------------------------------------------
// Hyperbolic Lorentz prototype-distance layer for MI455X (gfx1250, wave32).
//
// logits = DELTA * (1 - arccosh(max(t_i*yt_j - s_i*(A_i . Y_j), 1+eps)))
//   s_i = sinh(clip(||A_i||, eps, SINH_MAX)) / max(||A_i||, eps)
//   t_i = sqrt(1 + s_i^2 ||A_i||^2)   (x_time),  yt_j = sqrt(1 + ||Y_j||^2)
//
// GEMM: V_WMMA_F32_16X16X32_BF16, split-bf16 (A.B ~= Ah.Bh + Ah.Bl + Al.Bh).
// B operand staged into LDS with the Tensor Data Mover (tensor_load_to_lds),
// double-buffered in 64-wide K chunks, synchronized with s_wait_tensorcnt.
// ---------------------------------------------------------------------------

typedef __attribute__((ext_vector_type(16))) __bf16 v16bf;
typedef __attribute__((ext_vector_type(8)))  __bf16 v8bf;
typedef __attribute__((ext_vector_type(8)))  float  v8f;
typedef __attribute__((ext_vector_type(4)))  unsigned int u32x4;
typedef __attribute__((ext_vector_type(8)))  int    i32x8;
typedef __attribute__((ext_vector_type(4)))  int    i32x4;

#define BDIM      65536
#define DDIM      512
#define CDIM      1000
#define CPAD      1024
#define DELTA_F   1.4f
#define EPS_F     1e-8f
#define SINH_MAXF 11.09035488895912890625f  /* asinh(2^15) */

// LDS B-tile chunk: 64 rows (columns of output) x 64 K of bf16.
// TDM pads 16B after every 128B stored -> row stride 144B (72 bf16):
// 36-bank shift per row => 16 lanes x ds_load_b128 partition all 64 banks.
#define BROWS      64
#define KCHUNK     64
#define ROWSTRIDE  72        /* bf16 elements: 128B data + 16B pad */

static __device__ inline v8f wmma_bf16(v16bf a, v16bf b, v8f c) {
  return __builtin_amdgcn_wmma_f32_16x16x32_bf16(
      false, a, false, b, (short)0, c, false, false);
}

// Tensor DMA: load a 64x64 bf16 tile (row stride 512 elements) from a
// 1024x512 bf16 tensor into LDS with 16B padding per 128B row.
// D# built per CDNA5 ISA.
// This toolchain's builtin is the 6-arg form:
//   (uint32x4 g0, int32x8 g1, int32x4 g2, int32x4 g3, int32x8 extra, i32 cpol)
static __device__ inline void tdm_load_tile(unsigned int lds_addr,
                                            const __bf16* gsrc) {
  unsigned long long ga = (unsigned long long)gsrc;
  u32x4 g0;
  g0.x = 1u;                                   // count=1, no gather
  g0.y = lds_addr;                             // LDS byte address
  g0.z = (unsigned int)ga;                     // global_addr[31:0]
  g0.w = (unsigned int)(ga >> 32) | (2u << 30);// global_addr[56:32] | type=2

  i32x8 g1;
  g1[0] = (1 << 16)        // data_size = 1 -> 2 bytes
        | (1 << 20)        // pad_enable
        | (4 << 22)        // pad_interval = 4 -> every 32 DWORDs (128B)
        | (3 << 25);       // pad_amount   = 3 -> 4 DWORDs (16B)
  g1[1] = (int)(512u << 16);                   // tensor_dim0 = 512 (lo16)
  g1[2] = (int)(1024u << 16);                  // dim0 hi=0 | tensor_dim1=1024
  g1[3] = (int)(64u << 16);                    // dim1 hi=0 | tile_dim0 = 64
  g1[4] = 64;                                  // tile_dim1 = 64, tile_dim2 = 0
  g1[5] = 512;                                 // tensor_dim0_stride = 512
  g1[6] = 0;                                   // stride0 hi | dim1_stride lo
  g1[7] = 0;                                   // dim1_stride hi

  i32x4 g2;                                    // benign 3D/4D dims
  g2[0] = 1; g2[1] = 1; g2[2] = 0; g2[3] = 0;  // tensor_dim2=1, tensor_dim3=1
  i32x4 g3;
  g3[0] = 0; g3[1] = (1 << 16); g3[2] = 0; g3[3] = 0;  // tensor_dim4=1

  i32x8 gx = {0, 0, 0, 0, 0, 0, 0, 0};         // unused trailing group

  __builtin_amdgcn_tensor_load_to_lds(g0, g1, g2, g3, gx, 0);
}

static __device__ inline v16bf ld_frag16(const __bf16* p) {
  v8bf a = *(const v8bf*)p;
  v8bf b = *(const v8bf*)(p + 16);
  return __builtin_shufflevector(a, b, 0, 1, 2, 3, 4, 5, 6, 7,
                                 8, 9, 10, 11, 12, 13, 14, 15);
}

// ---------------------------------------------------------------------------
// Prep: cls_embedding (1000x512 f32) -> bf16 hi/lo (padded to 1024 rows,
// zero-filled) + y_time[1024].
// ---------------------------------------------------------------------------
__global__ void prep_cls_kernel(const float* __restrict__ cls,
                                __bf16* __restrict__ bhi,
                                __bf16* __restrict__ blo,
                                float* __restrict__ yt) {
  const int j   = blockIdx.x;   // 0..1023
  const int tid = threadIdx.x;  // 0..127
  float ssq = 0.f;
  for (int i = tid; i < DDIM; i += 128) {
    float x = (j < CDIM) ? cls[(size_t)j * DDIM + i] : 0.f;
    __bf16 h = (__bf16)x;
    bhi[(size_t)j * DDIM + i] = h;
    blo[(size_t)j * DDIM + i] = (__bf16)(x - (float)h);
    ssq += x * x;
  }
  __shared__ float red[128];
  red[tid] = ssq;
  __syncthreads();
  for (int s = 64; s > 0; s >>= 1) {
    if (tid < s) red[tid] += red[tid + s];
    __syncthreads();
  }
  if (tid == 0) yt[j] = sqrtf(1.f + red[0]);
}

// ---------------------------------------------------------------------------
// GEMM + epilogue. Block: 256 threads = 8 waves; wave w owns rows
// [bx*128 + 16w, +16) x cols [by*64, +64) as four 16x16 WMMA tiles.
// B tile double-buffered in LDS via TDM.
// ---------------------------------------------------------------------------
__global__ __launch_bounds__(256)
void hyper_dist_kernel(const float*  __restrict__ A,    // 65536 x 512 f32
                       const __bf16* __restrict__ Bhi,  // 1024 x 512
                       const __bf16* __restrict__ Blo,  // 1024 x 512
                       const float*  __restrict__ yt,   // 1024
                       float*        __restrict__ out)  // 65536 x 1000
{
  __shared__ __bf16 sB[2][2][BROWS * ROWSTRIDE];  // [buf][hi/lo][row*72+k]

  const int lane  = threadIdx.x & 31;
  const int wv    = threadIdx.x >> 5;
  const int halfL = lane >> 4;   // K-half selector in fragment layouts
  const int l16   = lane & 15;
  const int m0    = blockIdx.x * 128 + wv * 16;
  const int n0    = blockIdx.y * 64;

  // A fragment source: lane holds row (m0 + l16); per k-step it supplies
  // K-groups [32k + 8*halfL, +8) and [32k + 16 + 8*halfL, +8)  (ISA 7.12.2).
  const float* arow = A + (size_t)(m0 + l16) * DDIM + 8 * halfL;

  // Global chunk base for this block's B tile (rows n0..n0+63).
  const __bf16* gBhi = Bhi + (size_t)n0 * DDIM;
  const __bf16* gBlo = Blo + (size_t)n0 * DDIM;

  v8f acc0 = {}, acc1 = {}, acc2 = {}, acc3 = {};
  float ssq = 0.f;  // this lane's partial sum of squares of row (m0+l16)

  // ---- preload chunk 0 via Tensor Data Mover ----
  if (wv == 0) {
    tdm_load_tile((unsigned int)(size_t)&sB[0][0][0], gBhi);
    tdm_load_tile((unsigned int)(size_t)&sB[0][1][0], gBlo);
    __builtin_amdgcn_s_wait_tensorcnt(0);
  }
  __syncthreads();

  for (int c = 0; c < 8; ++c) {        // 8 chunks x 64 K
    // Issue DMA for next chunk into the other buffer (that buffer's readers
    // all passed the barrier at the end of iteration c-1).
    if (c < 7 && wv == 0) {
      tdm_load_tile((unsigned int)(size_t)&sB[(c + 1) & 1][0][0],
                    gBhi + (c + 1) * KCHUNK);
      tdm_load_tile((unsigned int)(size_t)&sB[(c + 1) & 1][1][0],
                    gBlo + (c + 1) * KCHUNK);
    }
    if (c < 7)
      __builtin_prefetch(arow + (c + 1) * KCHUNK, 0, 0);  // global_prefetch

    const __bf16* sh = &sB[c & 1][0][0];
    const __bf16* sl = &sB[c & 1][1][0];

#pragma unroll
    for (int k2 = 0; k2 < 2; ++k2) {   // two 32-K WMMA steps per chunk
      const int kk = c * 2 + k2;

      // ---- A fragment: 16 f32 -> bf16 hi/lo split + sum(x^2) ----
      const float4 a0 = *(const float4*)(arow + kk * 32);
      const float4 a1 = *(const float4*)(arow + kk * 32 + 4);
      const float4 a2 = *(const float4*)(arow + kk * 32 + 16);
      const float4 a3 = *(const float4*)(arow + kk * 32 + 20);
      float av[16] = {a0.x, a0.y, a0.z, a0.w, a1.x, a1.y, a1.z, a1.w,
                      a2.x, a2.y, a2.z, a2.w, a3.x, a3.y, a3.z, a3.w};
      v16bf ahi, alo;
#pragma unroll
      for (int e = 0; e < 16; ++e) {
        float x = av[e];
        __bf16 h = (__bf16)x;
        ahi[e] = h;
        alo[e] = (__bf16)(x - (float)h);
        ssq += x * x;
      }

      // ---- B fragments from LDS: lane = column (16*sub + l16),
      //      K = [32*k2 + 8*halfL, +8) and [32*k2 + 16 + 8*halfL, +8) within the chunk row ----
      const int koff = k2 * 32 + 8 * halfL;
      {
        const int r = (0 * 16 + l16) * ROWSTRIDE + koff;
        v16bf bh = ld_frag16(sh + r);
        v16bf bl = ld_frag16(sl + r);
        acc0 = wmma_bf16(ahi, bh, acc0);
        acc0 = wmma_bf16(ahi, bl, acc0);
        acc0 = wmma_bf16(alo, bh, acc0);
      }
      {
        const int r = (1 * 16 + l16) * ROWSTRIDE + koff;
        v16bf bh = ld_frag16(sh + r);
        v16bf bl = ld_frag16(sl + r);
        acc1 = wmma_bf16(ahi, bh, acc1);
        acc1 = wmma_bf16(ahi, bl, acc1);
        acc1 = wmma_bf16(alo, bh, acc1);
      }
      {
        const int r = (2 * 16 + l16) * ROWSTRIDE + koff;
        v16bf bh = ld_frag16(sh + r);
        v16bf bl = ld_frag16(sl + r);
        acc2 = wmma_bf16(ahi, bh, acc2);
        acc2 = wmma_bf16(ahi, bl, acc2);
        acc2 = wmma_bf16(alo, bh, acc2);
      }
      {
        const int r = (3 * 16 + l16) * ROWSTRIDE + koff;
        v16bf bh = ld_frag16(sh + r);
        v16bf bl = ld_frag16(sl + r);
        acc3 = wmma_bf16(ahi, bh, acc3);
        acc3 = wmma_bf16(ahi, bl, acc3);
        acc3 = wmma_bf16(alo, bh, acc3);
      }
    }

    if (c < 7) {
      if (wv == 0) __builtin_amdgcn_s_wait_tensorcnt(0);
      __syncthreads();
    }
  }

  // ---- per-row hyperbolic scalars (free: reuse of the A stream) ----
  float ssf = ssq + __shfl_xor(ssq, 16);
  float rc  = sqrtf(ssf);                              // curv = 1
  float si  = fminf(fmaxf(rc, EPS_F), SINH_MAXF);      // clip(sinh input)
  float s   = sinhf(si) / fmaxf(rc, EPS_F);            // exp-map row scale
  float t   = sqrtf(1.f + s * s * ssf);                // x_time

  // Accumulator element v lives on D-row (v + 8*halfL); lane (v + 8*halfL)
  // holds that row's scalars.
  float sr[8], tr[8];
#pragma unroll
  for (int v = 0; v < 8; ++v) {
    int src = v + 8 * halfL;
    sr[v] = __shfl(s, src, 32);
    tr[v] = __shfl(t, src, 32);
  }

  // ---- epilogue: Lorentz inner product -> arccosh -> logits ----
  const v8f accs[4] = {acc0, acc1, acc2, acc3};
#pragma unroll
  for (int sub = 0; sub < 4; ++sub) {
    int n = n0 + sub * 16 + l16;      // D-matrix column = lane % 16
    float ytn = yt[n];
    if (n < CDIM) {
#pragma unroll
      for (int v = 0; v < 8; ++v) {
        float dot = accs[sub][v];
        float c   = tr[v] * ytn - sr[v] * dot;       // -xyl  (curv = 1)
        float cm  = fmaxf(c, 1.f + EPS_F);
        float d   = logf(cm + sqrtf(cm * cm - 1.f)); // arccosh
        size_t m  = (size_t)(m0 + v + 8 * halfL);
        out[m * CDIM + n] = DELTA_F * (1.f - d);
      }
    }
  }
}

// ---------------------------------------------------------------------------
// Host launcher. d_in[0]=input_x (65536x512 f32), d_in[1]=cls (1000x512 f32).
// d_out: 65536x1000 f32. d_ws: ~2.1 MB used (B hi/lo padded + y_time).
// ---------------------------------------------------------------------------
extern "C" void kernel_launch(void* const* d_in, const int* in_sizes, int n_in,
                              void* d_out, int out_size, void* d_ws, size_t ws_size,
                              hipStream_t stream) {
  const float* input_x = (const float*)d_in[0];
  const float* cls     = (const float*)d_in[1];
  float* out           = (float*)d_out;

  char* w = (char*)d_ws;
  __bf16* bhi = (__bf16*)(w);
  __bf16* blo = (__bf16*)(w + (size_t)CPAD * DDIM * sizeof(__bf16));
  float*  yt  = (float*)(w + (size_t)CPAD * DDIM * sizeof(__bf16) * 2);

  prep_cls_kernel<<<dim3(CPAD), dim3(128), 0, stream>>>(cls, bhi, blo, yt);

  dim3 grid(BDIM / 128, CPAD / 64);   // 512 x 16
  hyper_dist_kernel<<<grid, dim3(256), 0, stream>>>(input_x, bhi, blo, yt, out);
}
